// TransformerBlock_57982058496475
// MI455X (gfx1250) — matrix operand order repaired and checked by reading, not compiled, not run on hardware
//
#include <hip/hip_runtime.h>
#include <math.h>

#define HW    65536
#define IMG   256
#define DIMC  192
#define HEADS 4
#define HD    48
#define HID   510
#define CH3   576     // 3*DIM
#define CH2   1020    // 2*HID
#define KSPLIT 32
#define EPS_LN 1e-6f

typedef __bf16 bf16;
typedef __attribute__((ext_vector_type(16))) __bf16 v16bf;
typedef __attribute__((ext_vector_type(8)))  float  v8f;

// A-matrix (16x32 bf16) per-lane K index for element e (ISA 7.12.2):
// lanes 0-15: K = {0..7,16..23}; lanes 16-31: K = {8..15,24..31}
__device__ __forceinline__ int a_kof(int lane, int e) {
  return ((lane & 16) >> 1) + ((e & 8) << 1) + (e & 7);
}

// XOR-swizzled LDS index for B panel stored [n][k] in 16B (8-elem) chunks.
// Keeps ds_load_b128 fragment reads bank-spread with zero padding.
__device__ __forceinline__ int bs_idx(int n, int k, int kstg) {
  int chunk = (k >> 3) ^ (n & 7);
  return n * kstg + chunk * 8 + (k & 7);
}

// ---------------------------------------------------------------- LN stats
__global__ __launch_bounds__(256)
void stats_partial(const float* __restrict__ x, float* __restrict__ part,
                   long perBatch, int blocksPerBatch) {
  int b = blockIdx.y;
  const float* xb = x + (size_t)b * perBatch;
  float s = 0.f, s2 = 0.f;
  for (long i = (long)blockIdx.x * 256 + threadIdx.x; i < perBatch;
       i += (long)blocksPerBatch * 256) {
    float v = xb[i]; s += v; s2 += v * v;
  }
  __shared__ float sh0[256], sh1[256];
  sh0[threadIdx.x] = s; sh1[threadIdx.x] = s2;
  __syncthreads();
  for (int o = 128; o > 0; o >>= 1) {
    if (threadIdx.x < o) {
      sh0[threadIdx.x] += sh0[threadIdx.x + o];
      sh1[threadIdx.x] += sh1[threadIdx.x + o];
    }
    __syncthreads();
  }
  if (threadIdx.x == 0) {
    part[(b * blocksPerBatch + blockIdx.x) * 2 + 0] = sh0[0];
    part[(b * blocksPerBatch + blockIdx.x) * 2 + 1] = sh1[0];
  }
}

__global__ void stats_final(const float* __restrict__ part, float* __restrict__ stats,
                            int blocksPerBatch, float n) {
  int b = threadIdx.x;
  if (b >= 2) return;
  float s = 0.f, s2 = 0.f;
  for (int i = 0; i < blocksPerBatch; ++i) {
    s  += part[(b * blocksPerBatch + i) * 2 + 0];
    s2 += part[(b * blocksPerBatch + i) * 2 + 1];
  }
  float mean = s / n;
  float var  = fmaxf((s2 - n * mean * mean) / (n - 1.f), 0.f);
  stats[b * 2 + 0] = mean;
  stats[b * 2 + 1] = 1.f / (sqrtf(var) + EPS_LN);  // eps added to std, per ref
}

// ---------------------------------------------------------------- generic WMMA GEMM
// C[M,N] = W[M,K](f32,row-major) x B[K,N], per batch (blockIdx.z).
// Whole K x 64 B-panel staged in LDS ONCE per block (activations hit HBM 1x),
// then all M tiles are looped in-block; weight fragments read from L2.
// BMODE 0: B = LayerNorm(f32 src) fused; BMODE 1: B = raw bf16.
// OMODE 0: f32 out + residual;     OMODE 1: bf16 out.
// KSTG: K rounded up to multiple of 32 (compile-time, sizes the LDS panel).
template <int BMODE, int OMODE, int KSTG>
__global__ __launch_bounds__(256)
void gemm_wmma(const float* __restrict__ W, const void* __restrict__ Bsrc,
               void* __restrict__ Out, const float* __restrict__ resid,
               const float* __restrict__ stats, const float* __restrict__ lnw,
               const float* __restrict__ lnb, int M, int K,
               long bStrideB, long bStrideC, long bStrideR) {
  __shared__ bf16 Bs[64 * KSTG];
  const int tid = threadIdx.x, lane = tid & 31, wave = tid >> 5;
  const int batch = blockIdx.z;
  const int n0 = blockIdx.x * 64;
  const int N = HW;

  // ---- stage full B panel (LN fused for BMODE 0), zero-fill K..KSTG ----
  {
    float mean = 0.f, istd = 1.f;
    const float* Bf = nullptr; const bf16* Bb = nullptr;
    if constexpr (BMODE == 0) {
      mean = stats[batch * 2]; istd = stats[batch * 2 + 1];
      Bf = (const float*)Bsrc + (size_t)batch * bStrideB;
    } else {
      Bb = (const bf16*)Bsrc + (size_t)batch * bStrideB;
    }
    for (int idx = tid; idx < 64 * KSTG; idx += 256) {
      int k = idx >> 6, n = idx & 63;
      float v = 0.f;
      if (k < K) {
        size_t gi = (size_t)k * N + (n0 + n);
        if constexpr (BMODE == 0) v = (Bf[gi] - mean) * istd * lnw[k] + lnb[k];
        else                      v = (float)Bb[gi];
      }
      Bs[bs_idx(n, k, KSTG)] = (bf16)v;
    }
  }
  __syncthreads();

  const int msub = wave & 3;     // 4 M-subtiles
  const int ngrp = wave >> 2;    // 2 N-groups of 2 subtiles
  const int mrow = msub * 16 + (lane & 15);

  for (int mt = 0; mt < M; mt += 64) {
    v8f acc0 = {}; v8f acc1 = {};
    const int gm = mt + mrow;
    const bool rok = gm < M;
    const float* wrow = W + (size_t)(rok ? gm : 0) * K;

    for (int k0 = 0; k0 < KSTG; k0 += 32) {
      v16bf af;
      #pragma unroll
      for (int e = 0; e < 16; ++e) {
        int gk = k0 + (lane & 16) + e;
        af[e] = (rok && gk < K) ? (bf16)wrow[gk] : (bf16)0.f;
      }
      int kb = k0 + (lane & 16);
      #pragma unroll
      for (int j = 0; j < 2; ++j) {
        v16bf bfv;
        int ncol = (ngrp * 2 + j) * 16 + (lane & 15);
        #pragma unroll
        for (int e = 0; e < 16; ++e) bfv[e] = Bs[bs_idx(ncol, kb + e, KSTG)];
        if (j == 0)
          acc0 = __builtin_amdgcn_wmma_f32_16x16x32_bf16(false, af, false, bfv,
                                                         (short)0, acc0, false, false);
        else
          acc1 = __builtin_amdgcn_wmma_f32_16x16x32_bf16(false, af, false, bfv,
                                                         (short)0, acc1, false, false);
      }
    }

    // epilogue (C/D: lanes 0-15 -> M=r, lanes 16-31 -> M=8+r; N = lane&15)
    #pragma unroll
    for (int j = 0; j < 2; ++j) {
      v8f a = j ? acc1 : acc0;
      int col   = n0 + (ngrp * 2 + j) * 16 + (lane & 15);
      int rbase = mt + msub * 16 + ((lane & 16) >> 1);
      #pragma unroll
      for (int r = 0; r < 8; ++r) {
        int rowg = rbase + r;
        if (rowg < M) {
          size_t oi = (size_t)rowg * N + col;
          if constexpr (OMODE == 0) {
            ((float*)Out)[(size_t)batch * bStrideC + oi] =
                a[r] + resid[(size_t)batch * bStrideR + oi];
          } else {
            ((bf16*)Out)[(size_t)batch * bStrideC + oi] = (bf16)a[r];
          }
        }
      }
    }
  }
}

// ---------------------------------------------------------------- depthwise 3x3 (SAME, zero pad)
__global__ __launch_bounds__(256)
void dwconv3_bf16(const bf16* __restrict__ in, const float* __restrict__ w,
                  bf16* __restrict__ out, int CH, long total) {
  long idx = (long)blockIdx.x * 256 + threadIdx.x;
  if (idx >= total) return;
  int pix = (int)(idx & (HW - 1));
  long bc = idx >> 16;
  int c = (int)(bc % CH);
  int y = pix >> 8, x = pix & 255;
  const bf16* base = in + (size_t)bc * HW;
  const float* w9 = w + (size_t)c * 9;
  float s = 0.f;
  #pragma unroll
  for (int dy = -1; dy <= 1; ++dy) {
    int yy = y + dy;
    if ((unsigned)yy >= IMG) continue;
    #pragma unroll
    for (int dx = -1; dx <= 1; ++dx) {
      int xx = x + dx;
      if ((unsigned)xx < IMG)
        s += w9[(dy + 1) * 3 + (dx + 1)] * (float)base[yy * IMG + xx];
    }
  }
  out[idx] = (bf16)s;
}

// ---------------------------------------------------------------- q,k row L2 norms
__global__ __launch_bounds__(256)
void rownorm(const bf16* __restrict__ qk, float* __restrict__ invnorm) {
  int rid = blockIdx.x;                 // 0..767  (b*384 + ch; q rows then k rows)
  int b = rid / 384, ch = rid % 384;
  const bf16* row = qk + ((size_t)b * CH3 + ch) * HW;
  float s = 0.f;
  for (int i = threadIdx.x; i < HW; i += 256) {
    float v = (float)row[i]; s += v * v;
  }
  __shared__ float sh[256];
  sh[threadIdx.x] = s; __syncthreads();
  for (int o = 128; o > 0; o >>= 1) {
    if (threadIdx.x < o) sh[threadIdx.x] += sh[threadIdx.x + o];
    __syncthreads();
  }
  if (threadIdx.x == 0)
    invnorm[rid] = 1.f / fmaxf(sqrtf(sh[0]), 1e-12f);
}

// ---------------------------------------------------------------- Gram: q @ k^T, split-K WMMA
__global__ __launch_bounds__(288)
void gram_wmma(const bf16* __restrict__ qk, float* __restrict__ apart) {
  int bh = blockIdx.y;                   // b*4 + h
  int b = bh >> 2, h = bh & 3;
  int wave = threadIdx.x >> 5, lane = threadIdx.x & 31;
  int mi = wave / 3, ni = wave % 3;      // 9 waves -> 3x3 tiles of 16
  const bf16* qrow = qk + ((size_t)b * CH3 + h * HD + mi * 16 + (lane & 15)) * HW;
  const bf16* krow = qk + ((size_t)b * CH3 + DIMC + h * HD + ni * 16 + (lane & 15)) * HW;
  v8f acc = {};
  int nb0 = blockIdx.x * (HW / KSPLIT);  // 2048 per split
  #pragma unroll 2
  for (int i = 0; i < (HW / KSPLIT) / 32; ++i) {
    int nb = nb0 + i * 32;
    v16bf af, bfv;
    #pragma unroll
    for (int e = 0; e < 16; ++e) af[e] = qrow[nb + (lane & 16) + e];
    int kb = nb + (lane & 16);
    #pragma unroll
    for (int e = 0; e < 16; ++e) bfv[e] = krow[kb + e];
    acc = __builtin_amdgcn_wmma_f32_16x16x32_bf16(false, af, false, bfv,
                                                  (short)0, acc, false, false);
  }
  int row = mi * 16 + ((lane & 16) >> 1);
  int col = ni * 16 + (lane & 15);
  float* dst = apart + (size_t)blockIdx.x * (2 * HEADS * HD * HD);
  #pragma unroll
  for (int r = 0; r < 8; ++r)
    dst[((size_t)bh * HD + row + r) * HD + col] = acc[r];
}

// ---------------------------------------------------------------- softmax (reduces split-K partials)
__global__ void softmax_attn(const float* __restrict__ apart,
                             const float* __restrict__ invnorm,
                             const float* __restrict__ temp,
                             bf16* __restrict__ out) {
  int rid = blockIdx.x * blockDim.x + threadIdx.x;   // b*192 + h*48 + c
  if (rid >= 2 * HEADS * HD) return;
  int b = rid / (HEADS * HD), rem = rid % (HEADS * HD);
  int h = rem / HD, c = rem % HD;
  float iq = invnorm[b * 384 + h * HD + c];
  float t  = temp[h];
  float l[HD];
  float mx = -1e30f;
  for (int d = 0; d < HD; ++d) {
    float raw = 0.f;
    for (int s = 0; s < KSPLIT; ++s)
      raw += apart[(size_t)s * (2 * HEADS * HD * HD) + (size_t)rid * HD + d];
    float v = raw * iq * invnorm[b * 384 + DIMC + h * HD + d] * t;
    l[d] = v; mx = fmaxf(mx, v);
  }
  float sum = 0.f;
  for (int d = 0; d < HD; ++d) { float e = expf(l[d] - mx); l[d] = e; sum += e; }
  float inv = 1.f / sum;
  bf16* o = out + (size_t)rid * HD;
  for (int d = 0; d < HD; ++d) o[d] = (bf16)(l[d] * inv);
}

// ---------------------------------------------------------------- attn @ v (K=48 padded to 64)
__global__ __launch_bounds__(96)
void attnv_wmma(const bf16* __restrict__ attn, const bf16* __restrict__ qk,
                bf16* __restrict__ out) {
  int bh = blockIdx.y; int b = bh >> 2, h = bh & 3;
  int wave = threadIdx.x >> 5, lane = threadIdx.x & 31;
  int n0 = blockIdx.x * 16;
  const bf16* ap = attn + (size_t)bh * HD * HD;
  const bf16* vbase = qk + ((size_t)b * CH3 + 2 * DIMC + h * HD) * HW;
  v8f acc = {};
  #pragma unroll
  for (int kc = 0; kc < 2; ++kc) {
    v16bf af, bfv;
    int row = wave * 16 + (lane & 15);
    #pragma unroll
    for (int e = 0; e < 16; ++e) {
      int k = kc * 32 + (lane & 16) + e;
      af[e] = (k < HD) ? ap[row * HD + k] : (bf16)0.f;
    }
    int col = n0 + (lane & 15);
    #pragma unroll
    for (int e = 0; e < 16; ++e) {
      int d = kc * 32 + (lane & 16) + e;
      bfv[e] = (d < HD) ? vbase[(size_t)d * HW + col] : (bf16)0.f;
    }
    acc = __builtin_amdgcn_wmma_f32_16x16x32_bf16(false, af, false, bfv,
                                                  (short)0, acc, false, false);
  }
  int ch  = h * HD + wave * 16 + ((lane & 16) >> 1);
  int col = n0 + (lane & 15);
  #pragma unroll
  for (int r = 0; r < 8; ++r)
    out[((size_t)b * DIMC + ch + r) * HW + col] = (bf16)acc[r];
}

// ---------------------------------------------------------------- fused dwconv + GELU gate
__global__ __launch_bounds__(256)
void ffdw_gate(const bf16* __restrict__ in, const float* __restrict__ w,
               bf16* __restrict__ out) {
  long total = (long)2 * HID * HW;
  long idx = (long)blockIdx.x * 256 + threadIdx.x;
  if (idx >= total) return;
  int pix = (int)(idx & (HW - 1));
  long bc = idx >> 16;
  int c = (int)(bc % HID);
  int b = (int)(bc / HID);
  int y = pix >> 8, x = pix & 255;
  const bf16* b1 = in + ((size_t)b * CH2 + c) * HW;
  const bf16* b2 = in + ((size_t)b * CH2 + HID + c) * HW;
  const float* w1 = w + (size_t)c * 9;
  const float* w2 = w + (size_t)(HID + c) * 9;
  float a = 0.f, g = 0.f;
  #pragma unroll
  for (int dy = -1; dy <= 1; ++dy) {
    int yy = y + dy;
    if ((unsigned)yy >= IMG) continue;
    #pragma unroll
    for (int dx = -1; dx <= 1; ++dx) {
      int xx = x + dx;
      if ((unsigned)xx < IMG) {
        int o = yy * IMG + xx, wi = (dy + 1) * 3 + (dx + 1);
        a += w1[wi] * (float)b1[o];
        g += w2[wi] * (float)b2[o];
      }
    }
  }
  float ge = 0.5f * a * (1.f + erff(a * 0.70710678118654752f));  // exact GELU
  out[idx] = (bf16)(ge * g);
}

// ---------------------------------------------------------------- host launcher
extern "C" void kernel_launch(void* const* d_in, const int* in_sizes, int n_in,
                              void* d_out, int out_size, void* d_ws, size_t ws_size,
                              hipStream_t stream) {
  (void)in_sizes; (void)n_in; (void)out_size; (void)ws_size;
  const float* x      = (const float*)d_in[0];
  const float* n1w    = (const float*)d_in[1];
  const float* n1b    = (const float*)d_in[2];
  const float* qkvw   = (const float*)d_in[3];
  const float* qkvdw  = (const float*)d_in[4];
  const float* projw  = (const float*)d_in[5];
  const float* temp   = (const float*)d_in[6];
  const float* n2w    = (const float*)d_in[7];
  const float* n2b    = (const float*)d_in[8];
  const float* ffinw  = (const float*)d_in[9];
  const float* ffdww  = (const float*)d_in[10];
  const float* ffoutw = (const float*)d_in[11];
  float* out = (float*)d_out;

  // workspace layout
  char* ws = (char*)d_ws;
  const size_t S_x2 = (size_t)2 * DIMC * HW * 4;        // 100 MB
  float* x2 = (float*)ws;
  char* small   = ws + S_x2;                            // 4 MB scratch
  float* stats1 = (float*)(small + 0);
  float* stats2 = (float*)(small + 64);
  float* part   = (float*)(small + 128);                // 2*256*2 f32
  float* invn   = (float*)(small + 8192);               // 2*384 f32
  bf16*  abf    = (bf16*)(small + 16384);               // 2*4*48*48 bf16
  float* apart  = (float*)(small + 65536);              // KSPLIT*18432 f32 (2.25MB)
  char* big = ws + S_x2 + (4u << 20);
  // phase A
  bf16* qkv_pre  = (bf16*)big;                          // 151 MB
  bf16* qkv_dwb  = (bf16*)(big + (size_t)2 * CH3 * HW * 2);
  bf16* attn_out = (bf16*)(big + (size_t)4 * CH3 * HW * 2);
  // phase B (aliases phase A, which is dead by then)
  bf16* ff_pre = (bf16*)big;                            // 267 MB
  bf16* gated  = (bf16*)(big + (size_t)2 * CH2 * HW * 2);

  const long perBatch = (long)DIMC * HW;

  // ---- MDTA ----
  stats_partial<<<dim3(256, 2), 256, 0, stream>>>(x, part, perBatch, 256);
  stats_final<<<1, 2, 0, stream>>>(part, stats1, 256, (float)perBatch);

  gemm_wmma<0, 1, 192><<<dim3(HW / 64, 1, 2), 256, 0, stream>>>(
      qkvw, x, qkv_pre, nullptr, stats1, n1w, n1b, CH3, DIMC,
      (long)DIMC * HW, (long)CH3 * HW, 0);

  long tdw = (long)2 * CH3 * HW;
  dwconv3_bf16<<<(tdw + 255) / 256, 256, 0, stream>>>(qkv_pre, qkvdw, qkv_dwb, CH3, tdw);

  rownorm<<<768, 256, 0, stream>>>(qkv_dwb, invn);
  gram_wmma<<<dim3(KSPLIT, 2 * HEADS), 288, 0, stream>>>(qkv_dwb, apart);
  softmax_attn<<<3, 128, 0, stream>>>(apart, invn, temp, abf);
  attnv_wmma<<<dim3(HW / 16, 2 * HEADS), 96, 0, stream>>>(abf, qkv_dwb, attn_out);

  gemm_wmma<1, 0, 192><<<dim3(HW / 64, 1, 2), 256, 0, stream>>>(
      projw, attn_out, x2, x, nullptr, nullptr, nullptr, DIMC, DIMC,
      (long)DIMC * HW, (long)DIMC * HW, (long)DIMC * HW);

  // ---- GDFN ----
  stats_partial<<<dim3(256, 2), 256, 0, stream>>>(x2, part, perBatch, 256);
  stats_final<<<1, 2, 0, stream>>>(part, stats2, 256, (float)perBatch);

  gemm_wmma<0, 1, 192><<<dim3(HW / 64, 1, 2), 256, 0, stream>>>(
      ffinw, x2, ff_pre, nullptr, stats2, n2w, n2b, CH2, DIMC,
      (long)DIMC * HW, (long)CH2 * HW, 0);

  long tg = (long)2 * HID * HW;
  ffdw_gate<<<(tg + 255) / 256, 256, 0, stream>>>(ff_pre, ffdww, gated);

  gemm_wmma<1, 0, 512><<<dim3(HW / 64, 1, 2), 256, 0, stream>>>(
      ffoutw, gated, out, x2, nullptr, nullptr, nullptr, DIMC, HID,
      (long)HID * HW, (long)DIMC * HW, (long)DIMC * HW);
}
